// ModelGuideHead_6528350290044
// MI455X (gfx1250) — compile-verified, matrix operand order checked by reading, not run on hardware
//
#include <hip/hip_runtime.h>
#include <hip/hip_bf16.h>

#define Bb 32
#define Ss 320
#define Dd 1024
#define NEGV (-1.0e9f)
#define EPSV (1.0e-10f)
#define REGV (0.1f)
#define NITER 300

typedef _Float16 v16h __attribute__((ext_vector_type(16)));
typedef _Float16 v8h  __attribute__((ext_vector_type(8)));
typedef float    v8f  __attribute__((ext_vector_type(8)));

// ---------------------------------------------------------------------------
// K1: per-token L2 norm + f16 normalized copy.  One block per (side,b,s).
// ---------------------------------------------------------------------------
__global__ void k_norm(const float* __restrict__ hs, const float* __restrict__ ht,
                       _Float16* __restrict__ hnS, _Float16* __restrict__ hnT,
                       float* __restrict__ norms) {
  int blk  = blockIdx.x;                 // side*B*S + b*S + s
  int side = blk / (Bb * Ss);
  int rem  = blk % (Bb * Ss);
  const float* h  = (side ? ht : hs) + (size_t)rem * Dd;
  _Float16*    hn = (side ? hnT : hnS) + (size_t)rem * Dd;
  int t = threadIdx.x;                   // 256 threads, 4 elems each
  float4 x = ((const float4*)h)[t];
  float ss = x.x * x.x + x.y * x.y + x.z * x.z + x.w * x.w;
  for (int o = 16; o; o >>= 1) ss += __shfl_xor(ss, o, 32);
  __shared__ float part[8];
  __shared__ float bc;
  if ((t & 31) == 0) part[t >> 5] = ss;
  __syncthreads();
  if (t == 0) {
    float tot = 0.f;
    for (int w = 0; w < 8; w++) tot += part[w];
    float nrm = sqrtf(tot);
    norms[blk] = nrm;
    bc = 1.0f / fmaxf(nrm, 1e-12f);
  }
  __syncthreads();
  float inv = bc;
  hn[t * 4 + 0] = (_Float16)(x.x * inv);
  hn[t * 4 + 1] = (_Float16)(x.y * inv);
  hn[t * 4 + 2] = (_Float16)(x.z * inv);
  hn[t * 4 + 3] = (_Float16)(x.w * inv);
}

// ---------------------------------------------------------------------------
// K2: masks + log-marginals.  One block per (side,b), 320 threads.
// ---------------------------------------------------------------------------
__global__ void k_marg(const int* __restrict__ tokS, const int* __restrict__ tokT,
                       const float* __restrict__ norms,
                       float* __restrict__ mask, float* __restrict__ logm) {
  int side = blockIdx.x >> 5;
  int b    = blockIdx.x & 31;
  int s    = threadIdx.x;
  const int* tok = (side ? tokT : tokS) + b * Ss;
  int idx = side * Bb * Ss + b * Ss + s;
  int tk  = tok[s];
  float mv = (tk == 0 || tk == 101 || tk == 102) ? 0.f : 1.f;
  float nm = norms[idx] * mv;
  float w = nm;
  for (int o = 16; o; o >>= 1) w += __shfl_xor(w, o, 32);
  __shared__ float part[10];
  __shared__ float tot;
  if ((s & 31) == 0) part[s >> 5] = w;
  __syncthreads();
  if (s == 0) {
    float tt = 0.f;
    for (int i = 0; i < 10; i++) tt += part[i];
    tot = tt;
  }
  __syncthreads();
  float a = nm / tot;
  mask[idx] = mv;
  logm[idx] = (mv > 0.f) ? __logf(fmaxf(a, 1e-38f)) : NEGV;
}

// ---------------------------------------------------------------------------
// K3: cosine-distance matrix via WMMA.  One 16x16 tile per wave.
// d[b,i,j] = (1 - cos)/2
// ---------------------------------------------------------------------------
__global__ void k_gemm(const _Float16* __restrict__ A, const _Float16* __restrict__ Bt,
                       float* __restrict__ dmat) {
  int wave = (int)((blockIdx.x * blockDim.x + threadIdx.x) >> 5);
  int lane = threadIdx.x & 31;
  int b  = wave / 400;
  int r  = wave % 400;
  int i0 = (r / 20) * 16;
  int j0 = (r % 20) * 16;
  const _Float16* Ab = A  + (size_t)b * Ss * Dd;
  const _Float16* Bc = Bt + (size_t)b * Ss * Dd;
  int half = lane >> 4;                 // lanes 16-31 hold the K+8 halves
  int lr   = lane & 15;
  const _Float16* arow = Ab + (size_t)(i0 + lr) * Dd + half * 8;
  const _Float16* brow = Bc + (size_t)(j0 + lr) * Dd + half * 8;
  v8f c = {0.f, 0.f, 0.f, 0.f, 0.f, 0.f, 0.f, 0.f};
  for (int k0 = 0; k0 < Dd; k0 += 32) {
    union { v16h v; v8h h[2]; } a, bb;
    a.h[0]  = *(const v8h*)(arow + k0);
    a.h[1]  = *(const v8h*)(arow + k0 + 16);
    bb.h[0] = *(const v8h*)(brow + k0);
    bb.h[1] = *(const v8h*)(brow + k0 + 16);
    c = __builtin_amdgcn_wmma_f32_16x16x32_f16(false, a.v, false, bb.v,
                                               (short)0, c, false, false);
  }
  float* drow = dmat + (size_t)b * Ss * Ss + (size_t)(i0 + half * 8) * Ss + (j0 + lr);
#pragma unroll
  for (int rr = 0; rr < 8; rr++) drow[(size_t)rr * Ss] = (1.0f - c[rr]) * 0.5f;
}

// ---------------------------------------------------------------------------
// K4: per-row min/max over valid targets -> affine constants for M0.
// M0[i][j] = d*c0[i] + o0[i]  (valid entries)
// ---------------------------------------------------------------------------
__global__ void k_rowstat(const float* __restrict__ dmat, const float* __restrict__ mask,
                          float* __restrict__ c0, float* __restrict__ o0) {
  int blk = blockIdx.x;                 // b*S + i
  int b = blk / Ss;
  int i = blk % Ss;
  const float* row = dmat + (size_t)b * Ss * Ss + (size_t)i * Ss;
  const float* mt  = mask + Bb * Ss + b * Ss;
  int t = threadIdx.x;                  // 128 threads
  float mn = 2.0f, mx = -1.0f;
  for (int j = t; j < Ss; j += 128) {
    float d = row[j];
    bool  v = mt[j] > 0.f;
    mn = fminf(mn, v ? d : 2.0f);
    mx = fmaxf(mx, v ? d : -1.0f);
  }
  for (int o = 16; o; o >>= 1) {
    mn = fminf(mn, __shfl_xor(mn, o, 32));
    mx = fmaxf(mx, __shfl_xor(mx, o, 32));
  }
  __shared__ float pn[4], px[4];
  if ((t & 31) == 0) { pn[t >> 5] = mn; px[t >> 5] = mx; }
  __syncthreads();
  if (t == 0) {
    for (int w = 1; w < 4; w++) { mn = fminf(mn, pn[w]); mx = fmaxf(mx, px[w]); }
    float inv = 1.0f / (mx - mn + EPSV);
    c0[blk] = -inv / REGV;
    o0[blk] = (mn - EPSV) * inv / REGV;
  }
}

// K5: per-column min/max over valid sources -> affine constants for M1.
__global__ void k_colstat(const float* __restrict__ dmat, const float* __restrict__ mask,
                          float* __restrict__ c1, float* __restrict__ o1) {
  int b = blockIdx.x;
  int j = threadIdx.x;                  // 320 threads: coalesced column sweep
  const float* dm = dmat + (size_t)b * Ss * Ss;
  const float* ms = mask + b * Ss;
  float mn = 2.f, mx = -1.f;
  for (int i = 0; i < Ss; i++) {
    float d = dm[(size_t)i * Ss + j];
    bool  v = ms[i] > 0.f;
    mn = fminf(mn, v ? d : 2.f);
    mx = fmaxf(mx, v ? d : -1.f);
  }
  float inv = 1.f / (mx - mn + EPSV);
  c1[b * Ss + j] = -inv / REGV;
  o1[b * Ss + j] = (mn - EPSV) * inv / REGV;
}

// ---------------------------------------------------------------------------
// K6: E[m][i][j] = exp(M[m][i][j]); exactly 0 at masked entries.
// M in [-10,0] so E in [e^-10, 1]: perfectly representable in f32.
// ---------------------------------------------------------------------------
__global__ void k_expm(const float* __restrict__ dmat, const float* __restrict__ mask,
                       const float* __restrict__ c0, const float* __restrict__ o0,
                       const float* __restrict__ c1, const float* __restrict__ o1,
                       float* __restrict__ Emat) {
  size_t gid = (size_t)blockIdx.x * 256 + threadIdx.x;
  const size_t per = (size_t)Ss * Ss;
  int    m = (int)(gid / per);
  size_t e = gid % per;
  int i = (int)(e / Ss);
  int j = (int)(e % Ss);
  int kk = m >> 5;
  int b  = m & 31;
  float r = 0.f;
  if ((mask[b * Ss + i] > 0.f) && (mask[Bb * Ss + b * Ss + j] > 0.f)) {
    float d    = dmat[(size_t)b * per + e];
    float base = kk ? (d * c1[b * Ss + j] + o1[b * Ss + j])
                    : (d * c0[b * Ss + i] + o0[b * Ss + i]);
    r = __expf(base);
  }
  Emat[gid] = r;
}

// ---------------------------------------------------------------------------
// K7: Sinkhorn with precomputed E: each half-iteration is a GEMV,
//   lse_j = log(sum_i E[i][j]*exp(u_i - umax)) + umax
// One block (10 waves) per matrix; pure FMA stream out of L2.
// ---------------------------------------------------------------------------
__global__ void __launch_bounds__(320)
k_sinkhorn(const float* __restrict__ E_, const float* __restrict__ logm,
           float* __restrict__ u_g, float* __restrict__ v_g,
           float* __restrict__ tmn_g, float* __restrict__ tmx_g) {
  int m = blockIdx.x;                   // m = k*32 + b
  int b = m & 31;
  const float* E = E_ + (size_t)m * Ss * Ss;
  __shared__ float u[Ss], v[Ss], la[Ss], lb[Ss], eu[Ss], ev[Ss];
  __shared__ float pr[10], pn[10], px[10];
  __shared__ float bcast;
  int tid  = threadIdx.x;
  int lane = tid & 31;
  int wid  = tid >> 5;
  la[tid] = logm[b * Ss + tid];
  lb[tid] = logm[Bb * Ss + b * Ss + tid];
  u[tid] = 0.f;
  v[tid] = 0.f;
  __syncthreads();

  for (int it = 0; it < NITER; ++it) {
    // ---- umax + eu ----
    {
      float x = u[tid];
      for (int o = 16; o; o >>= 1) x = fmaxf(x, __shfl_xor(x, o, 32));
      if (lane == 0) pr[wid] = x;
      __syncthreads();
      if (tid == 0) {
        float mxx = pr[0];
        for (int w = 1; w < 10; w++) mxx = fmaxf(mxx, pr[w]);
        bcast = mxx;
      }
      __syncthreads();
      eu[tid] = __expf(u[tid] - bcast);
    }
    float umax = bcast;
    __syncthreads();
    // ---- v-update: thread = column j, 4 accumulators, coalesced sweep ----
    {
      int j = tid;
      const float* col = E + j;
      float a0 = 0.f, a1 = 0.f, a2 = 0.f, a3 = 0.f;
      for (int i = 0; i < Ss; i += 4) {
        a0 += col[(size_t)(i + 0) * Ss] * eu[i + 0];
        a1 += col[(size_t)(i + 1) * Ss] * eu[i + 1];
        a2 += col[(size_t)(i + 2) * Ss] * eu[i + 2];
        a3 += col[(size_t)(i + 3) * Ss] * eu[i + 3];
      }
      float acc = (a0 + a1) + (a2 + a3);
      // masked column -> acc==0 -> v=0; value provably unused downstream
      v[j] = (acc > 0.f) ? (lb[j] - (__logf(acc) + umax)) : 0.f;
    }
    __syncthreads();
    // ---- vmax + ev ----
    {
      float x = v[tid];
      for (int o = 16; o; o >>= 1) x = fmaxf(x, __shfl_xor(x, o, 32));
      if (lane == 0) pr[wid] = x;
      __syncthreads();
      if (tid == 0) {
        float mxx = pr[0];
        for (int w = 1; w < 10; w++) mxx = fmaxf(mxx, pr[w]);
        bcast = mxx;
      }
      __syncthreads();
      ev[tid] = __expf(v[tid] - bcast);
    }
    float vmax = bcast;
    __syncthreads();
    // ---- u-update: wave per row, lane add-reduce ----
    for (int r = 0; r < 32; r++) {
      int i = wid * 32 + r;
      const float* rowp = E + (size_t)i * Ss;
      float a = 0.f;
#pragma unroll
      for (int q = 0; q < 10; q++) {
        int j = lane + q * 32;
        a += rowp[j] * ev[j];
      }
      for (int o = 16; o; o >>= 1) a += __shfl_xor(a, o, 32);
      if (lane == 0) u[i] = (a > 0.f) ? (la[i] - (__logf(a) + vmax)) : 0.f;
    }
    __syncthreads();
  }

  // ---- epilogue: per-matrix min/max of T = E*exp(u+v) over valid entries ----
  {
    int j = tid;
    float mn = 1e9f, mxv = -1.0f;
    if (lb[j] > -1e8f) {
      float vj = v[j];
      for (int i = 0; i < Ss; i++) {
        if (la[i] <= -1e8f) continue;   // uniform branch across the block
        float T = E[(size_t)i * Ss + j] * __expf(u[i] + vj);
        mn  = fminf(mn, T);
        mxv = fmaxf(mxv, T);
      }
    }
    for (int o = 16; o; o >>= 1) {
      mn  = fminf(mn, __shfl_xor(mn, o, 32));
      mxv = fmaxf(mxv, __shfl_xor(mxv, o, 32));
    }
    if (lane == 0) { pn[wid] = mn; px[wid] = mxv; }
    __syncthreads();
    if (tid == 0) {
      for (int w = 1; w < 10; w++) { mn = fminf(mn, pn[w]); mxv = fmaxf(mxv, px[w]); }
      tmn_g[m] = mn;
      tmx_g[m] = mxv;
    }
    u_g[m * Ss + tid] = u[tid];
    v_g[m * Ss + tid] = v[tid];
  }
}

// ---------------------------------------------------------------------------
// K8: normalized transport plan + BCE-sum, reduced into d_out[0].
// ---------------------------------------------------------------------------
__global__ void k_bce(const float* __restrict__ Emat, const float* __restrict__ mask,
                      const float* __restrict__ u_g, const float* __restrict__ v_g,
                      const float* __restrict__ tmn_g, const float* __restrict__ tmx_g,
                      const float* __restrict__ guide, float* __restrict__ out) {
  size_t gid = (size_t)blockIdx.x * 256 + threadIdx.x;   // < 2*32*320*320
  const size_t per = (size_t)Ss * Ss;
  int    m = (int)(gid / per);
  size_t e = gid % per;
  int i = (int)(e / Ss);
  int j = (int)(e % Ss);
  int b = m & 31;
  bool valid = (mask[b * Ss + i] > 0.f) && (mask[Bb * Ss + b * Ss + j] > 0.f);
  float p = 0.f;
  if (valid) {
    float T = Emat[gid] * __expf(u_g[m * Ss + i] + v_g[m * Ss + j]);
    p = (T - tmn_g[m] + EPSV) / (tmx_g[m] - tmn_g[m] + EPSV);
  }
  float g   = guide[(size_t)b * per + e];
  float lp  = (p > 0.f) ? __logf(fmaxf(p, 1e-38f)) : -100.f;
  float l1p = (p < 1.f) ? __logf(fmaxf(1.f - p, 1e-38f)) : -100.f;
  float term = g * lp + (1.f - g) * l1p;
  for (int o = 16; o; o >>= 1) term += __shfl_xor(term, o, 32);
  __shared__ float part[8];
  int t = threadIdx.x;
  if ((t & 31) == 0) part[t >> 5] = term;
  __syncthreads();
  if (t == 0) {
    float tot = 0.f;
    for (int w = 0; w < 8; w++) tot += part[w];
    atomicAdd(out, tot * (-1.0f / (float)Bb));
  }
}

// ---------------------------------------------------------------------------
extern "C" void kernel_launch(void* const* d_in, const int* in_sizes, int n_in,
                              void* d_out, int out_size, void* d_ws, size_t ws_size,
                              hipStream_t stream) {
  const float* hs    = (const float*)d_in[0];
  const float* ht    = (const float*)d_in[1];
  const float* guide = (const float*)d_in[2];
  const int*   tokS  = (const int*)d_in[3];
  const int*   tokT  = (const int*)d_in[4];
  float* out = (float*)d_out;

  char* base = (char*)d_ws;
  size_t off = 0;
  auto carve = [&](size_t bytes) -> char* {
    char* p = base + off;
    off += (bytes + 255) & ~(size_t)255;
    return p;
  };
  // region0: hn (f16 staging, dead after k_gemm) aliased with E (exp(M)).
  const size_t hnBytes = (size_t)2 * Bb * Ss * Dd;          // 20,971,520 each
  const size_t eBytes  = (size_t)4 * 64 * Ss * Ss;          // 26,214,400
  size_t r0 = 2 * hnBytes;                                  // 41.9 MB >= eBytes
  char* region0 = carve(r0);
  _Float16* hnS  = (_Float16*)region0;
  _Float16* hnT  = (_Float16*)(region0 + hnBytes);
  float*    Emat = (float*)region0;     // written after gemm (stream-ordered)
  (void)eBytes;

  float* norms = (float*)carve((size_t)4 * 2 * Bb * Ss);
  float* maskA = (float*)carve((size_t)4 * 2 * Bb * Ss);
  float* logm  = (float*)carve((size_t)4 * 2 * Bb * Ss);
  float* dmat  = (float*)carve((size_t)4 * Bb * Ss * Ss);
  float* c0    = (float*)carve((size_t)4 * Bb * Ss);
  float* o0    = (float*)carve((size_t)4 * Bb * Ss);
  float* c1    = (float*)carve((size_t)4 * Bb * Ss);
  float* o1    = (float*)carve((size_t)4 * Bb * Ss);
  float* u_g   = (float*)carve((size_t)4 * 64 * Ss);
  float* v_g   = (float*)carve((size_t)4 * 64 * Ss);
  float* tmn   = (float*)carve((size_t)4 * 64);
  float* tmx   = (float*)carve((size_t)4 * 64);

  hipMemsetAsync(d_out, 0, sizeof(float), stream);

  k_norm<<<2 * Bb * Ss, 256, 0, stream>>>(hs, ht, hnS, hnT, norms);
  k_marg<<<2 * Bb, Ss, 0, stream>>>(tokS, tokT, norms, maskA, logm);
  k_gemm<<<(Bb * 20 * 20) / 8, 256, 0, stream>>>(hnS, hnT, dmat);
  k_rowstat<<<Bb * Ss, 128, 0, stream>>>(dmat, maskA, c0, o0);
  k_colstat<<<Bb, Ss, 0, stream>>>(dmat, maskA, c1, o1);
  k_expm<<<(2 * Bb * Ss * Ss) / 256, 256, 0, stream>>>(dmat, maskA, c0, o0, c1, o1, Emat);
  k_sinkhorn<<<64, Ss, 0, stream>>>(Emat, logm, u_g, v_g, tmn, tmx);
  k_bce<<<(2 * Bb * Ss * Ss) / 256, 256, 0, stream>>>(Emat, maskA, u_g, v_g,
                                                      tmn, tmx, guide, out);
}
